// EclipseAwareMultiHeadAttention_9199819948535
// MI455X (gfx1250) — matrix operand order repaired and checked by reading, not compiled, not run on hardware
//
#include <hip/hip_runtime.h>

// ---------------- problem constants ----------------
#define BB 4
#define SS 1024
#define DD 1024
#define HH 16
#define DK 64
#define MM (BB * SS)          // 4096 rows for the projection GEMMs
#define SCALE 0.125f          // 1/sqrt(64)
#define NEGV -1.0e9f

typedef __attribute__((ext_vector_type(16))) __bf16 v16bf;
typedef __attribute__((ext_vector_type(8)))  __bf16 v8bf;
typedef __attribute__((ext_vector_type(8)))  float  v8f;
typedef __attribute__((ext_vector_type(4)))  unsigned int v4u;
typedef __attribute__((ext_vector_type(8)))  int v8i;
typedef __attribute__((ext_vector_type(4)))  int v4i;

#if defined(__has_builtin)
#if __has_builtin(__builtin_amdgcn_tensor_load_to_lds) && __has_builtin(__builtin_amdgcn_s_wait_tensorcnt)
#define HAVE_TDM 1
#endif
#endif
#ifndef HAVE_TDM
#define HAVE_TDM 0
#endif

static __device__ __forceinline__ __bf16 f2bf(float f) {
    unsigned u = __builtin_bit_cast(unsigned, f);
    unsigned r = u + 0x7FFFu + ((u >> 16) & 1u);   // round-to-nearest-even
    unsigned short h = (unsigned short)(r >> 16);
    return __builtin_bit_cast(__bf16, h);
}

#if HAVE_TDM
// ---- Tensor Data Mover: 2-D tile (2-byte elements) global -> LDS ----
// D# layout per CDNA5 ISA 8.3/8.4: group0 = {flags, lds_addr, global_addr, type},
// group1 = {data_size, tensor dims, tile dims, dim0 stride}. 2-D tensor: groups 2/3
// unused (zero). This toolchain exposes the 6-arg builtin:
//   (uint32x4 g0, int32x8 g1, int32x4, int32x4, int32x8, i32 cpol)
static __device__ __forceinline__ void tdm_load_2d(
    unsigned lds_off, const void* gaddr,
    unsigned tensor_d0, unsigned tensor_d1,
    unsigned tile_d0, unsigned tile_d1, unsigned stride_d0_elems) {
    unsigned long long ga = (unsigned long long)(uintptr_t)gaddr;
    unsigned ga_lo = (unsigned)__builtin_amdgcn_readfirstlane((int)(ga & 0xFFFFFFFFull));
    unsigned ga_hi = (unsigned)__builtin_amdgcn_readfirstlane((int)((ga >> 32) & 0xFFFFFFFFull));
    unsigned lds_u = (unsigned)__builtin_amdgcn_readfirstlane((int)lds_off);

    v4u g0;
    g0[0] = 1u;                                        // count=1, user-mode load
    g0[1] = lds_u;                                     // lds_addr (bytes)
    g0[2] = ga_lo;                                     // global_addr[31:0]
    g0[3] = (ga_hi & 0x1FFFFFFu) | (2u << 30);         // global_addr[56:32] | type=2

    v8i g1;
    g1[0] = (int)(1u << 16);                           // workgroup_mask=0, data_size=1 (2B)
    g1[1] = (int)((tensor_d0 & 0xFFFFu) << 16);        // barrier_addr=0 | tensor_dim0[15:0]
    g1[2] = (int)(((tensor_d0 >> 16) & 0xFFFFu) | ((tensor_d1 & 0xFFFFu) << 16));
    g1[3] = (int)(((tensor_d1 >> 16) & 0xFFFFu) | ((tile_d0 & 0xFFFFu) << 16));
    g1[4] = (int)(tile_d1 & 0xFFFFu);                  // tile_dim1 | tile_dim2=0
    g1[5] = (int)stride_d0_elems;                      // tensor_dim0_stride[31:0]
    g1[6] = 0;                                         // stride0 hi | dim1_stride lo
    g1[7] = 0;

    v4i g2 = {0, 0, 0, 0};
    v4i g3 = {0, 0, 0, 0};
    v8i gx = {0, 0, 0, 0, 0, 0, 0, 0};
    __builtin_amdgcn_tensor_load_to_lds(g0, g1, g2, g3, gx, 0);
}
#endif

// ---------------- f32 -> bf16 conversion ----------------
__global__ void cvt_bf16_kernel(const float* __restrict__ src,
                                __bf16* __restrict__ dst, int n) {
    int i = blockIdx.x * blockDim.x + threadIdx.x;
    if (i < n) dst[i] = f2bf(src[i]);
}

// ---------------- GEMM: C[M,N] = A[M,K](bf16) * W[K,N](bf16) + bias ----------------
// block = 128 threads (4 waves). Block tile 64x64, each wave owns 16 rows x 64 cols.
// W tiles are DMA'd by the Tensor Data Mover into a double-buffered LDS staging area
// (overlapped with the WMMAs), then transposed in LDS for contiguous B-fragment reads.
__global__ __launch_bounds__(128) void gemm_bias_kernel(
    const __bf16* __restrict__ A, const __bf16* __restrict__ W,
    const float* __restrict__ bias,
    float* __restrict__ outF, __bf16* __restrict__ outB,
    int Mdim, int Ndim, int Kdim) {

    __shared__ __align__(16) __bf16 Wraw[2][32 * 64];  // row-major TDM landing buffers
    __shared__ __align__(16) __bf16 Wt[64 * 32];       // transposed: [n][k]

    const int tid  = threadIdx.x;
    const int lane = tid & 31;
    const int wave = tid >> 5;
    const int ln   = lane & 15;
    const int lh   = lane >> 4;
    const int n0   = blockIdx.x * 64;
    const int m0   = blockIdx.y * 64;
    const int rowA = m0 + wave * 16 + ln;
    const int kb   = lh * 8;

    v8f acc[4] = {};

#if HAVE_TDM
    int bufsel = 0;
    if (wave == 0) {
        unsigned lds0 = (unsigned)(uintptr_t)(void*)&Wraw[0][0];
        tdm_load_2d(lds0, W + (size_t)0 * Ndim + n0,
                    (unsigned)Ndim, (unsigned)Kdim, 64u, 32u, (unsigned)Ndim);
    }
#endif

    #pragma unroll 1
    for (int kk = 0; kk < Kdim; kk += 32) {
#if HAVE_TDM
        if (wave == 0) __builtin_amdgcn_s_wait_tensorcnt(0);
        __syncthreads();
        // ---- transpose Wraw[bufsel] (32 x 64) into Wt [n][k] ----
        {
            int r  = tid >> 2;          // 0..31 : k-row
            int cg = (tid & 3) * 16;    // column group
            const __bf16* wp = &Wraw[bufsel][r * 64 + cg];
            v8bf w0 = *(const v8bf*)(wp);
            v8bf w1 = *(const v8bf*)(wp + 8);
            #pragma unroll
            for (int i = 0; i < 8; ++i) {
                Wt[(cg + i)     * 32 + r] = w0[i];
                Wt[(cg + 8 + i) * 32 + r] = w1[i];
            }
        }
        __syncthreads();
        // ---- kick off TDM for the next W tile (overlaps with the WMMAs below) ----
        if (wave == 0 && (kk + 32) < Kdim) {
            unsigned ldsn = (unsigned)(uintptr_t)(void*)&Wraw[bufsel ^ 1][0];
            tdm_load_2d(ldsn, W + (size_t)(kk + 32) * Ndim + n0,
                        (unsigned)Ndim, (unsigned)Kdim, 64u, 32u, (unsigned)Ndim);
        }
        bufsel ^= 1;
#else
        // ---- fallback: stage W tile transposed straight from global ----
        {
            int r  = tid >> 2;
            int cg = (tid & 3) * 16;
            const __bf16* wp = W + (size_t)(kk + r) * Ndim + n0 + cg;
            v8bf w0 = *(const v8bf*)(wp);
            v8bf w1 = *(const v8bf*)(wp + 8);
            #pragma unroll
            for (int i = 0; i < 8; ++i) {
                Wt[(cg + i)     * 32 + r] = w0[i];
                Wt[(cg + 8 + i) * 32 + r] = w1[i];
            }
        }
        __syncthreads();
#endif

        // ---- A fragment (16-bit A layout: lane<16 K0..7/16..23, lane>=16 K8..15/24..31) ----
        const __bf16* ap = A + (size_t)rowA * Kdim + kk + kb;
        v8bf alo = *(const v8bf*)(ap);
        v8bf ahi = *(const v8bf*)(ap + 16);
        if (kk + 32 < Kdim) __builtin_prefetch(ap + 32, 0, 1);
        v16bf afrag;
        #pragma unroll
        for (int i = 0; i < 8; ++i) { afrag[i] = alo[i]; afrag[8 + i] = ahi[i]; }

        // ---- 4 WMMAs across the 64-wide N tile ----
        #pragma unroll
        for (int c = 0; c < 4; ++c) {
            const __bf16* bp = &Wt[(c * 16 + ln) * 32 + lh * 8];
            v8bf b0 = *(const v8bf*)(bp);
            v8bf b1 = *(const v8bf*)(bp + 16);
            v16bf bfrag;
            #pragma unroll
            for (int i = 0; i < 8; ++i) { bfrag[i] = b0[i]; bfrag[8 + i] = b1[i]; }
            acc[c] = __builtin_amdgcn_wmma_f32_16x16x32_bf16(
                false, afrag, false, bfrag, (short)0, acc[c], false, false);
        }
        __syncthreads();
    }

    // ---- epilogue: bias + store (f32 and/or bf16) ----
    #pragma unroll
    for (int c = 0; c < 4; ++c) {
        int col = n0 + c * 16 + ln;
        float bv = bias ? bias[col] : 0.0f;
        #pragma unroll
        for (int j = 0; j < 8; ++j) {
            int row = m0 + wave * 16 + j + lh * 8;
            float v = acc[c][j] + bv;
            if (outF) outF[(size_t)row * Ndim + col] = v;
            if (outB) outB[(size_t)row * Ndim + col] = f2bf(v);
        }
    }
}

// ---------------- attention pass: one wave per (b, h, 16-query tile) ----------------
// Flash-style online softmax; raw masked/boosted/scaled scores streamed to attn_raw;
// per-row running max/sum saved for the normalization pass; context (bf16) to ws.
// V tiles are TDM'd into LDS at the top of each iteration and the DMA is hidden
// behind the K-fragment loads, score WMMAs and softmax VALU work.
__global__ __launch_bounds__(32) void attn_pass_kernel(
    const __bf16* __restrict__ Qb, const __bf16* __restrict__ Kb,
    const __bf16* __restrict__ Vb,
    const unsigned char* __restrict__ eclipse, const int* __restrict__ maskp,
    float* __restrict__ attn_raw, __bf16* __restrict__ ctx,
    float* __restrict__ mbuf, float* __restrict__ lbuf) {

    __shared__ __align__(16) __bf16 Pls[16 * 16];   // P tile (q x key)
    __shared__ __align__(16) __bf16 Vraw[16 * 64];  // TDM landing buffer (key x dim)
    __shared__ __align__(16) __bf16 Vt[64 * 32];    // V transposed [dim][key], K padded to 32

    const int lane = threadIdx.x;
    const int qt = blockIdx.x;    // query tile   0..63
    const int h  = blockIdx.y;    // head         0..15
    const int b  = blockIdx.z;    // batch        0..3
    const int ln = lane & 15;
    const int lh = lane >> 4;
    const __bf16 bz = __builtin_bit_cast(__bf16, (unsigned short)0);

    // zero the padded K half of Vt once (A is zero there too; avoid NaN garbage)
    for (int i = lane; i < 64 * 16; i += 32) {
        Vt[(i >> 4) * 32 + 16 + (i & 15)] = bz;
    }

    // ---- Q fragments: 16 rows x 64 dims -> two 16x32 A fragments ----
    const int qrow = qt * 16 + ln;
    const __bf16* qp = Qb + ((size_t)(b * SS + qrow)) * DD + h * DK + lh * 8;
    v16bf a0, a1;
    {
        v8bf t0 = *(const v8bf*)(qp);
        v8bf t1 = *(const v8bf*)(qp + 16);
        v8bf t2 = *(const v8bf*)(qp + 32);
        v8bf t3 = *(const v8bf*)(qp + 48);
        #pragma unroll
        for (int i = 0; i < 8; ++i) {
            a0[i] = t0[i]; a0[8 + i] = t1[i];
            a1[i] = t2[i]; a1[8 + i] = t3[i];
        }
    }

    float m[8], l[8];
    #pragma unroll
    for (int j = 0; j < 8; ++j) { m[j] = -1.0e30f; l[j] = 0.0f; }
    v8f acc[4] = {};

#if HAVE_TDM
    const unsigned vraw_off = (unsigned)(uintptr_t)(void*)&Vraw[0];
#endif

    #pragma unroll 1
    for (int kt = 0; kt < SS / 16; ++kt) {
        const int key = kt * 16 + ln;

#if HAVE_TDM
        // ---- TDM: V tile (16 keys x 64 dims, row stride DD) -> LDS ----
        tdm_load_2d(vraw_off, Vb + ((size_t)(b * SS + kt * 16)) * DD + h * DK,
                    (unsigned)DD, (unsigned)(BB * SS), 64u, 16u, (unsigned)DD);
#endif

        // ---- scores: S = Q(16x64) * K_tile^T -> two WMMAs ----
        const __bf16* kp = Kb + ((size_t)(b * SS + kt * 16 + ln)) * DD + h * DK + lh * 8;
        v16bf kb0, kb1;
        {
            v8bf t0 = *(const v8bf*)(kp);
            v8bf t1 = *(const v8bf*)(kp + 16);
            v8bf t2 = *(const v8bf*)(kp + 32);
            v8bf t3 = *(const v8bf*)(kp + 48);
            #pragma unroll
            for (int i = 0; i < 8; ++i) {
                kb0[i] = t0[i]; kb0[8 + i] = t1[i];
                kb1[i] = t2[i]; kb1[8 + i] = t3[i];
            }
        }
        v8f s = {};
        s = __builtin_amdgcn_wmma_f32_16x16x32_bf16(false, a0, false, kb0, (short)0, s, false, false);
        s = __builtin_amdgcn_wmma_f32_16x16x32_bf16(false, a1, false, kb1, (short)0, s, false, false);

        const float bst = (eclipse[(size_t)b * SS + key] ? 2.0f : 1.0f) * SCALE;

        v8f p;
        #pragma unroll
        for (int j = 0; j < 8; ++j) {
            int q = qt * 16 + j + lh * 8;
            float sv = s[j] * bst;
            if (maskp[((size_t)b * SS + q) * SS + key] == 0) sv = NEGV;
            attn_raw[(((size_t)(b * HH + h)) * SS + q) * SS + key] = sv;

            // online softmax row update (reduce across the 16 key lanes)
            float tmax = sv;
            #pragma unroll
            for (int off = 8; off; off >>= 1) tmax = fmaxf(tmax, __shfl_xor(tmax, off, 16));
            float mnew  = fmaxf(m[j], tmax);
            float alpha = __expf(m[j] - mnew);
            float pv    = __expf(sv - mnew);
            float psum  = pv;
            #pragma unroll
            for (int off = 8; off; off >>= 1) psum += __shfl_xor(psum, off, 16);
            l[j] = l[j] * alpha + psum;
            m[j] = mnew;
            p[j] = pv;
            #pragma unroll
            for (int c = 0; c < 4; ++c) acc[c][j] *= alpha;
        }

        // ---- P -> LDS (bf16, 16x16) ----
        #pragma unroll
        for (int j = 0; j < 8; ++j) Pls[(j + lh * 8) * 16 + ln] = f2bf(p[j]);

        // ---- V tile -> transposed LDS [dim][key] ----
        {
            int keyl = lane >> 1;
            int dh   = (lane & 1) * 32;
#if HAVE_TDM
            __builtin_amdgcn_s_wait_tensorcnt(0);
            const __bf16* vp = &Vraw[keyl * 64 + dh];
#else
            const __bf16* vp = Vb + ((size_t)(b * SS + kt * 16 + keyl)) * DD + h * DK + dh;
#endif
            v8bf v0 = *(const v8bf*)(vp);
            v8bf v1 = *(const v8bf*)(vp + 8);
            v8bf v2 = *(const v8bf*)(vp + 16);
            v8bf v3 = *(const v8bf*)(vp + 24);
            #pragma unroll
            for (int i = 0; i < 8; ++i) {
                Vt[(dh + i)      * 32 + keyl] = v0[i];
                Vt[(dh + 8 + i)  * 32 + keyl] = v1[i];
                Vt[(dh + 16 + i) * 32 + keyl] = v2[i];
                Vt[(dh + 24 + i) * 32 + keyl] = v3[i];
            }
        }
        __syncthreads();

        // ---- A fragment from P (real K=16, upper 16 zero-padded) ----
        v16bf pa;
        {
            const __bf16* pp = &Pls[ln * 16 + lh * 8];
            v8bf p0 = *(const v8bf*)(pp);
            #pragma unroll
            for (int i = 0; i < 8; ++i) { pa[i] = p0[i]; pa[8 + i] = bz; }
        }

        // ---- context += P * V : 4 WMMAs across 64 dims ----
        #pragma unroll
        for (int c = 0; c < 4; ++c) {
            const __bf16* bp = &Vt[(c * 16 + ln) * 32 + lh * 8];
            v8bf b0 = *(const v8bf*)(bp);
            v8bf b1 = *(const v8bf*)(bp + 16);
            v16bf vb;
            #pragma unroll
            for (int i = 0; i < 8; ++i) { vb[i] = b0[i]; vb[8 + i] = b1[i]; }
            acc[c] = __builtin_amdgcn_wmma_f32_16x16x32_bf16(
                false, pa, false, vb, (short)0, acc[c], false, false);
        }
        __syncthreads();
    }

    // ---- epilogue: normalize context, store bf16; save m/l per row ----
    #pragma unroll
    for (int j = 0; j < 8; ++j) {
        int q = qt * 16 + j + lh * 8;
        float inv = 1.0f / l[j];
        #pragma unroll
        for (int c = 0; c < 4; ++c) {
            ctx[((size_t)(b * SS + q)) * DD + h * DK + c * 16 + ln] = f2bf(acc[c][j] * inv);
        }
        if (ln == 0) {
            mbuf[((size_t)(b * HH + h)) * SS + q] = m[j];
            lbuf[((size_t)(b * HH + h)) * SS + q] = l[j];
        }
    }
}

// ---------------- normalize raw scores into probabilities ----------------
__global__ void attn_norm_kernel(float* __restrict__ attn,
                                 const float* __restrict__ mbuf,
                                 const float* __restrict__ lbuf, size_t n4) {
    size_t idx = (size_t)blockIdx.x * blockDim.x + threadIdx.x;
    if (idx >= n4) return;
    size_t base = idx * 4;
    size_t row  = base >> 10;                  // / SS
    float mrow = mbuf[row];
    float inv  = 1.0f / lbuf[row];
    float4 v = *(const float4*)(attn + base);
    v.x = __expf(v.x - mrow) * inv;
    v.y = __expf(v.y - mrow) * inv;
    v.z = __expf(v.z - mrow) * inv;
    v.w = __expf(v.w - mrow) * inv;
    *(float4*)(attn + base) = v;
}

// ---------------- host side ----------------
extern "C" void kernel_launch(void* const* d_in, const int* in_sizes, int n_in,
                              void* d_out, int out_size, void* d_ws, size_t ws_size,
                              hipStream_t stream) {
    const float* query = (const float*)d_in[0];
    const float* key_  = (const float*)d_in[1];
    const float* value = (const float*)d_in[2];
    const unsigned char* eclipse = (const unsigned char*)d_in[3];
    const int*   maskp = (const int*)d_in[4];
    const float* Wq = (const float*)d_in[5];
    const float* bq = (const float*)d_in[6];
    const float* Wk = (const float*)d_in[7];
    const float* bk = (const float*)d_in[8];
    const float* Wv = (const float*)d_in[9];
    const float* bv = (const float*)d_in[10];
    const float* Wo = (const float*)d_in[11];
    const float* bo = (const float*)d_in[12];

    float* out_f  = (float*)d_out;                               // [B,S,D]
    float* attn_f = out_f + (size_t)BB * SS * DD;                // [B,H,S,S]

    char* ws = (char*)d_ws;
    size_t off = 0;
    auto alloc = [&](size_t bytes) -> void* {
        void* p = ws + off;
        off += (bytes + 255) & ~(size_t)255;
        return p;
    };
    const size_t actB = (size_t)MM * DD * sizeof(__bf16);   // 8 MB
    const size_t wB   = (size_t)DD * DD * sizeof(__bf16);   // 2 MB
    __bf16* Xq  = (__bf16*)alloc(actB);
    __bf16* Xk  = (__bf16*)alloc(actB);
    __bf16* Xv  = (__bf16*)alloc(actB);
    __bf16* Wqb = (__bf16*)alloc(wB);
    __bf16* Wkb = (__bf16*)alloc(wB);
    __bf16* Wvb = (__bf16*)alloc(wB);
    __bf16* Wob = (__bf16*)alloc(wB);
    __bf16* Qb  = (__bf16*)alloc(actB);
    __bf16* Kb  = (__bf16*)alloc(actB);
    __bf16* Vb  = (__bf16*)alloc(actB);
    __bf16* Ctx = (__bf16*)alloc(actB);
    float*  mbuf = (float*)alloc((size_t)BB * HH * SS * sizeof(float));
    float*  lbuf = (float*)alloc((size_t)BB * HH * SS * sizeof(float));

    // 1) convert activations + weights to bf16
    const int nAct = MM * DD, nW = DD * DD;
    cvt_bf16_kernel<<<(nAct + 255) / 256, 256, 0, stream>>>(query, Xq, nAct);
    cvt_bf16_kernel<<<(nAct + 255) / 256, 256, 0, stream>>>(key_,  Xk, nAct);
    cvt_bf16_kernel<<<(nAct + 255) / 256, 256, 0, stream>>>(value, Xv, nAct);
    cvt_bf16_kernel<<<(nW + 255) / 256, 256, 0, stream>>>(Wq, Wqb, nW);
    cvt_bf16_kernel<<<(nW + 255) / 256, 256, 0, stream>>>(Wk, Wkb, nW);
    cvt_bf16_kernel<<<(nW + 255) / 256, 256, 0, stream>>>(Wv, Wvb, nW);
    cvt_bf16_kernel<<<(nW + 255) / 256, 256, 0, stream>>>(Wo, Wob, nW);

    // 2) Q/K/V projections (bf16 out)
    dim3 ggrid(DD / 64, MM / 64);
    gemm_bias_kernel<<<ggrid, 128, 0, stream>>>(Xq, Wqb, bq, nullptr, Qb, MM, DD, DD);
    gemm_bias_kernel<<<ggrid, 128, 0, stream>>>(Xk, Wkb, bk, nullptr, Kb, MM, DD, DD);
    gemm_bias_kernel<<<ggrid, 128, 0, stream>>>(Xv, Wvb, bv, nullptr, Vb, MM, DD, DD);

    // 3) attention (flash-style, raw scores to attn_f, context to Ctx)
    attn_pass_kernel<<<dim3(SS / 16, HH, BB), 32, 0, stream>>>(
        Qb, Kb, Vb, eclipse, maskp, attn_f, Ctx, mbuf, lbuf);

    // 4) normalize attn probabilities
    size_t n4 = (size_t)BB * HH * SS * SS / 4;
    attn_norm_kernel<<<(unsigned)((n4 + 255) / 256), 256, 0, stream>>>(attn_f, mbuf, lbuf, n4);

    // 5) output projection (f32 out)
    gemm_bias_kernel<<<ggrid, 128, 0, stream>>>(Ctx, Wob, bo, out_f, nullptr, MM, DD, DD);

    (void)in_sizes; (void)n_in; (void)out_size; (void)ws_size;
}
